// DecoderBlock_21449066676387
// MI455X (gfx1250) — compile-verified, matrix operand order checked by reading, not run on hardware
//
#include <hip/hip_runtime.h>
#include <hip/hip_bf16.h>

typedef __attribute__((ext_vector_type(16))) __bf16 v16bf;
typedef __attribute__((ext_vector_type(8)))  float  v8f;
typedef __attribute__((ext_vector_type(4)))  unsigned int uint4v;
typedef __attribute__((ext_vector_type(8)))  int    int8v;
typedef __attribute__((ext_vector_type(4)))  int    int4v;
typedef unsigned short u16;
typedef unsigned int   u32;

#if __has_builtin(__builtin_amdgcn_tensor_load_to_lds)
#define HAVE_TDM 1
#else
#define HAVE_TDM 0
#endif

static __device__ __forceinline__ u16 f2bf(float f) {
    u32 u = __float_as_uint(f);
    u32 r = (u + 0x7FFFu + ((u >> 16) & 1u)) >> 16;   // round-to-nearest-even
    return (u16)r;
}

union FragBF { u32 u[8]; v16bf v; };

// K-pair index for 16-bit A/B WMMA layouts: VGPR v holds K = kk, kk+1.
static __device__ __forceinline__ int kpair(int v, int half) {
    return ((v < 4) ? 0 : 16) + half * 8 + 2 * (v & 3);
}

// Generic->LDS offset: low 32 bits of a generic shared address are the LDS
// byte offset (aperture format, ISA 10.2).
static __device__ __forceinline__ u32 lds_off(const void* p) {
    return (u32)(size_t)p;
}

// Async DMA of 16B global -> LDS, tracked by ASYNCcnt (CDNA5).
static __device__ __forceinline__ void async_b128(void* lds_dst, const void* gsrc) {
    asm volatile("global_load_async_to_lds_b128 %0, %1, off"
                 :: "v"(lds_off(lds_dst)), "v"((unsigned long long)(size_t)gsrc)
                 : "memory");
}
static __device__ __forceinline__ void wait_async0() {
    asm volatile("s_wait_asynccnt 0x0" ::: "memory");
}

#if HAVE_TDM
// Tensor Data Mover: DMA a 2D tile (tile_d0 x tile_d1 elements, 2B each) from
// a row-major tensor into LDS (linear fill). D# layout per CDNA5 ISA ch.8.
// NOTE: this toolchain exposes the 6-arg builtin
//   (v4u32 g0, v8i32 g1, v4i32 g2, v4i32 g3, v8i32, i32 cpol)
static __device__ __forceinline__ void tdm_load_2d_bf16(
    void* lds_dst, const void* gsrc,
    unsigned tensor_d0, unsigned tensor_d1,
    unsigned tile_d0, unsigned tile_d1, unsigned stride_d0) {
    unsigned long long ga = (unsigned long long)(size_t)gsrc;
    uint4v g0;
    g0.x = 1u;                                         // count=1 (valid), user
    g0.y = lds_off(lds_dst);                           // lds_addr
    g0.z = (u32)(ga & 0xFFFFFFFFull);                  // global_addr[31:0]
    g0.w = (u32)((ga >> 32) & 0x01FFFFFFull)           // global_addr[56:32]
         | 0x80000000u;                                // type=2 ("image")
    int8v g1;
    g1[0] = (int)(1u << 16);                           // data_size=1 (2 bytes)
    g1[1] = (int)((tensor_d0 & 0xFFFFu) << 16);        // tensor_dim0 lo
    g1[2] = (int)((tensor_d0 >> 16) | ((tensor_d1 & 0xFFFFu) << 16));
    g1[3] = (int)((tensor_d1 >> 16) | (tile_d0 << 16));// tensor_dim1 hi | tile_dim0
    g1[4] = (int)tile_d1;                              // tile_dim1 (tile_dim2=0)
    g1[5] = (int)stride_d0;                            // tensor_dim0_stride lo
    g1[6] = 0;                                         // stride hi, dim1_stride lo
    g1[7] = 0;
    const int4v z4 = {0, 0, 0, 0};
    const int8v z8 = {0, 0, 0, 0, 0, 0, 0, 0};
    __builtin_amdgcn_tensor_load_to_lds(g0, g1, z4, z4, z8, 0);
}
#endif

// ---------------------------------------------------------------------------
// LayerNorm: one block per row, fp32 in -> bf16 out
// ---------------------------------------------------------------------------
__global__ __launch_bounds__(256) void ln_kernel(const float* __restrict__ x,
                                                 const float* __restrict__ g,
                                                 const float* __restrict__ b,
                                                 u16* __restrict__ out, int C) {
    __shared__ float red[256];
    const int row = blockIdx.x, tid = threadIdx.x;
    const float* xr = x + (size_t)row * C;
    float s = 0.f, s2 = 0.f;
    for (int i = tid; i < C; i += 256) { float v = xr[i]; s += v; s2 += v * v; }
    red[tid] = s; __syncthreads();
    for (int st = 128; st > 0; st >>= 1) { if (tid < st) red[tid] += red[tid + st]; __syncthreads(); }
    const float mean = red[0] / (float)C;
    __syncthreads();
    red[tid] = s2; __syncthreads();
    for (int st = 128; st > 0; st >>= 1) { if (tid < st) red[tid] += red[tid + st]; __syncthreads(); }
    const float var  = red[0] / (float)C - mean * mean;
    const float rstd = rsqrtf(var + 1e-5f);
    u16* orow = out + (size_t)row * C;
    for (int i = tid; i < C; i += 256)
        orow[i] = f2bf((xr[i] - mean) * rstd * g[i] + b[i]);
}

// ---------------------------------------------------------------------------
// fp32 -> bf16 convert
// ---------------------------------------------------------------------------
__global__ __launch_bounds__(256) void cvt_kernel(const float* __restrict__ in,
                                                  u16* __restrict__ out, long n) {
    long i = (long)blockIdx.x * 256 + threadIdx.x;
    if (i < n) out[i] = f2bf(in[i]);
}

// ---------------------------------------------------------------------------
// Tiled bf16 WMMA GEMM: out = act(A[M,K] * W[K,N] + bias) + res
// Block tile 256x64, BK=32, 8 waves; each wave: 32 rows x 64 cols = 8 WMMA
// accumulators. A-tile DMA'd by the Tensor Data Mover (wave 0 issues one
// descriptor per tile); W-tile converted f32->bf16 through VGPRs. LDS tiles
// are double-buffered so tile t+1 fill overlaps tile t WMMAs.
// ---------------------------------------------------------------------------
#define GM_BM 256
#define GM_BN 64
#define GM_BK 32

template <bool HAS_BIAS, bool GELU, bool HAS_RES, bool OUT_F, bool OUT_B>
__global__ __launch_bounds__(256) void gemm_bf16_kernel(
    const u16* __restrict__ A, const float* __restrict__ W,
    const float* __restrict__ bias, const float* __restrict__ res,
    float* __restrict__ outF, u16* __restrict__ outB,
    int M, int K, int N) {
    __shared__ __align__(16) u16 lA[2][GM_BM * GM_BK];   // [m][k]
    __shared__ __align__(16) u16 lW[2][GM_BN * GM_BK];   // [n][k]
    const int tid = threadIdx.x;
    const int wave = tid >> 5, lane = tid & 31, lid = lane & 15, half = lane >> 4;
    const int m0 = blockIdx.y * GM_BM;
    const int n0 = blockIdx.x * GM_BN;

    v8f acc[2][4];
    const v8f vzero = {0.f, 0.f, 0.f, 0.f, 0.f, 0.f, 0.f, 0.f};
    for (int ms = 0; ms < 2; ++ms)
        for (int nt = 0; nt < 4; ++nt) acc[ms][nt] = vzero;

    auto issueA = [&](int k0, int buf) {
#if HAVE_TDM
        if (wave == 0)
            tdm_load_2d_bf16(&lA[buf][0], &A[(size_t)m0 * K + k0],
                             (unsigned)K, (unsigned)M, GM_BK, GM_BM, (unsigned)K);
#else
        #pragma unroll
        for (int i = 0; i < 4; ++i) {
            int idx = i * 256 + tid;
            int r = idx >> 2, c8 = (idx & 3) * 8;
            async_b128(&lA[buf][r * GM_BK + c8],
                       &A[(size_t)(m0 + r) * K + k0 + c8]);
        }
#endif
    };
    auto waitA = [&]() {
#if HAVE_TDM
        if (wave == 0) __builtin_amdgcn_s_wait_tensorcnt(0);
#else
        wait_async0();
#endif
    };
    auto stageW = [&](int k0, int buf) {
        #pragma unroll
        for (int i = 0; i < 8; ++i) {
            int idx = i * 256 + tid;            // 2048 elems
            int kk = idx >> 6, nn = idx & 63;
            lW[buf][nn * GM_BK + kk] = f2bf(W[(size_t)(k0 + kk) * N + n0 + nn]);
        }
        if (k0 + GM_BK < K)
            __builtin_prefetch(&W[(size_t)(k0 + GM_BK) * N + n0 + (tid & 63)], 0, 3);
    };

    const int nk = K / GM_BK;
    issueA(0, 0);
    stageW(0, 0);
    int cur = 0;
    for (int t = 0; t < nk; ++t) {
        waitA();
        __syncthreads();                        // buf[cur] A+W ready, other buf free
        if (t + 1 < nk) {
            issueA((t + 1) * GM_BK, cur ^ 1);
            stageW((t + 1) * GM_BK, cur ^ 1);
        }
        FragBF fa[2];
        #pragma unroll
        for (int ms = 0; ms < 2; ++ms) {
            const u32* pa = (const u32*)&lA[cur][(wave * 32 + ms * 16 + lid) * GM_BK];
            #pragma unroll
            for (int v = 0; v < 8; ++v) fa[ms].u[v] = pa[kpair(v, half) >> 1];
        }
        #pragma unroll
        for (int nt = 0; nt < 4; ++nt) {
            FragBF fb;
            const u32* pb = (const u32*)&lW[cur][(nt * 16 + lid) * GM_BK];
            #pragma unroll
            for (int v = 0; v < 8; ++v) fb.u[v] = pb[kpair(v, half) >> 1];
            #pragma unroll
            for (int ms = 0; ms < 2; ++ms)
                acc[ms][nt] = __builtin_amdgcn_wmma_f32_16x16x32_bf16(
                    false, fa[ms].v, false, fb.v, (short)0, acc[ms][nt], false, false);
        }
        cur ^= 1;
    }

    // Epilogue: bias -> gelu -> residual; compile-time-selected outputs
    #pragma unroll
    for (int ms = 0; ms < 2; ++ms) {
        const int mrow0 = m0 + wave * 32 + ms * 16 + half * 8;
        #pragma unroll
        for (int nt = 0; nt < 4; ++nt) {
            const int col = n0 + nt * 16 + lid;
            float bv = 0.f;
            if (HAS_BIAS) bv = bias[col];
            #pragma unroll
            for (int v = 0; v < 8; ++v) {
                const size_t off = (size_t)(mrow0 + v) * N + col;
                float val = acc[ms][nt][v] + bv;
                if (GELU)    val = 0.5f * val * (1.f + erff(val * 0.70710678118f));
                if (HAS_RES) val += res[off];
                if (OUT_F)   outF[off] = val;
                if (OUT_B)   outB[off] = f2bf(val);
            }
        }
    }
}

// ---------------------------------------------------------------------------
// Flash attention (bf16 WMMA, fp32 online softmax).
// One wave per 16-row Q tile; 4 waves/block share K/V staging for one (b,h).
// K tile filled with async DMA; V staged transposed for packed B-fragments.
// ---------------------------------------------------------------------------
__global__ __launch_bounds__(128) void attn_kernel(
    const u16* __restrict__ qb, int qstride,
    const u16* __restrict__ kvb, int kvstride, int koff, int voff,
    const float* __restrict__ mask,
    u16* __restrict__ outB, int outStride,
    int H, int N) {
    __shared__ __align__(16) u16 lK[32 * 64];      // [key][d]
    __shared__ __align__(16) u16 lVt[64 * 32];     // [d][key]
    __shared__ __align__(16) u16 lP[4][16 * 32];   // per-wave probs [q][key]

    const int tid = threadIdx.x;
    const int wave = tid >> 5, lane = tid & 31, lid = lane & 15, half = lane >> 4;
    const int qtiles = N / 64;
    const int bh = blockIdx.x / qtiles;
    const int qt = blockIdx.x % qtiles;
    const int b = bh / H, h = bh % H;
    const int qrow0 = qt * 64 + wave * 16;

    const size_t qbase = (size_t)b * N * qstride + h * 64;
    const size_t kbase = (size_t)b * N * kvstride + koff + h * 64;
    const size_t vbase = (size_t)b * N * kvstride + voff + h * 64;

    const v8f vzero = {0.f, 0.f, 0.f, 0.f, 0.f, 0.f, 0.f, 0.f};
    v8f o[4];
    for (int t = 0; t < 4; ++t) o[t] = vzero;
    float mrow[8], lrow[8];
    #pragma unroll
    for (int v = 0; v < 8; ++v) { mrow[v] = -1e30f; lrow[v] = 0.f; }

    // Q fragments (A-layout) for d-chunks 0 and 32 — loaded once
    FragBF qa[2];
    #pragma unroll
    for (int dc = 0; dc < 2; ++dc)
        #pragma unroll
        for (int v = 0; v < 8; ++v) {
            int d = dc * 32 + kpair(v, half);
            qa[dc].u[v] = *(const u32*)&qb[qbase + (size_t)(qrow0 + lid) * qstride + d];
        }

    for (int c0 = 0; c0 < N; c0 += 32) {
        // Async DMA K tile (32 keys x 64 d): 256 x 16B chunks / 128 threads
        #pragma unroll
        for (int i = 0; i < 2; ++i) {
            int idx = i * 128 + tid;
            int key = idx >> 3, d8 = (idx & 7) * 8;
            async_b128(&lK[key * 64 + d8],
                       &kvb[kbase + (size_t)(c0 + key) * kvstride + d8]);
        }
        // Manual transposed V stage
        #pragma unroll
        for (int i = 0; i < 16; ++i) {
            int idx = i * 128 + tid;            // 2048 elems
            int key = idx >> 6, d = idx & 63;
            lVt[d * 32 + key] = kvb[vbase + (size_t)(c0 + key) * kvstride + d];
        }
        wait_async0();
        __syncthreads();

        // Scores: S[16 x 32] as two 16x16 tiles, reduce over D=64
        v8f s[2];
        #pragma unroll
        for (int t = 0; t < 2; ++t) {
            v8f z = vzero;
            #pragma unroll
            for (int dc = 0; dc < 2; ++dc) {
                FragBF fk;
                const u32* pk = (const u32*)&lK[(t * 16 + lid) * 64 + dc * 32];
                #pragma unroll
                for (int v = 0; v < 8; ++v) fk.u[v] = pk[kpair(v, half) >> 1];
                z = __builtin_amdgcn_wmma_f32_16x16x32_bf16(
                    false, qa[dc].v, false, fk.v, (short)0, z, false, false);
            }
            s[t] = z;
        }

        // Scale + additive mask (column = key)
        float mk0 = (1.f - mask[(size_t)b * N + c0 + lid])      * -5.f;
        float mk1 = (1.f - mask[(size_t)b * N + c0 + 16 + lid]) * -5.f;
        #pragma unroll
        for (int v = 0; v < 8; ++v) {
            s[0][v] = s[0][v] * 0.125f + mk0;
            s[1][v] = s[1][v] * 0.125f + mk1;
        }

        // Online softmax per row (row = half*8 + v, spread over 16 lanes of half)
        #pragma unroll
        for (int v = 0; v < 8; ++v) {
            float x = fmaxf(s[0][v], s[1][v]);
            x = fmaxf(x, __shfl_xor(x, 1, 32));
            x = fmaxf(x, __shfl_xor(x, 2, 32));
            x = fmaxf(x, __shfl_xor(x, 4, 32));
            x = fmaxf(x, __shfl_xor(x, 8, 32));
            const float mn = fmaxf(mrow[v], x);
            const float alpha = __expf(mrow[v] - mn);
            mrow[v] = mn;
            s[0][v] = __expf(s[0][v] - mn);
            s[1][v] = __expf(s[1][v] - mn);
            float sm = s[0][v] + s[1][v];
            sm += __shfl_xor(sm, 1, 32);
            sm += __shfl_xor(sm, 2, 32);
            sm += __shfl_xor(sm, 4, 32);
            sm += __shfl_xor(sm, 8, 32);
            lrow[v] = lrow[v] * alpha + sm;
            #pragma unroll
            for (int dt = 0; dt < 4; ++dt) o[dt][v] *= alpha;
            lP[wave][(half * 8 + v) * 32 + lid]      = f2bf(s[0][v]);
            lP[wave][(half * 8 + v) * 32 + 16 + lid] = f2bf(s[1][v]);
        }
        __syncthreads();

        // O += P[16x32] * V[32x64]  (4 d-tiles)
        FragBF pa;
        const u32* pp = (const u32*)&lP[wave][lid * 32];
        #pragma unroll
        for (int v = 0; v < 8; ++v) pa.u[v] = pp[kpair(v, half) >> 1];
        #pragma unroll
        for (int dt = 0; dt < 4; ++dt) {
            FragBF fv;
            const u32* pv = (const u32*)&lVt[(dt * 16 + lid) * 32];
            #pragma unroll
            for (int v = 0; v < 8; ++v) fv.u[v] = pv[kpair(v, half) >> 1];
            o[dt] = __builtin_amdgcn_wmma_f32_16x16x32_bf16(
                false, pa.v, false, fv.v, (short)0, o[dt], false, false);
        }
        __syncthreads();
    }

    // Normalize and store merged-head bf16 output
    #pragma unroll
    for (int v = 0; v < 8; ++v) {
        const float inv = 1.f / lrow[v];
        const size_t row = (size_t)b * N + qrow0 + half * 8 + v;
        #pragma unroll
        for (int dt = 0; dt < 4; ++dt) {
            const int col = h * 64 + dt * 16 + lid;
            outB[row * outStride + col] = f2bf(o[dt][v] * inv);
        }
    }
}

// ---------------------------------------------------------------------------
// Host-side orchestration
// ---------------------------------------------------------------------------
extern "C" void kernel_launch(void* const* d_in, const int* in_sizes, int n_in,
                              void* d_out, int out_size, void* d_ws, size_t ws_size,
                              hipStream_t stream) {
    (void)in_sizes; (void)n_in; (void)out_size; (void)ws_size;
    const float* x     = (const float*)d_in[0];
    const float* enc   = (const float*)d_in[1];
    const float* mask  = (const float*)d_in[2];
    const float* emask = (const float*)d_in[3];
    const float* ln1_g = (const float*)d_in[4];
    const float* ln1_b = (const float*)d_in[5];
    const float* qkv_w = (const float*)d_in[6];
    const float* sa_pw = (const float*)d_in[7];
    const float* sa_pb = (const float*)d_in[8];
    const float* ln2_g = (const float*)d_in[9];
    const float* ln2_b = (const float*)d_in[10];
    const float* q_w   = (const float*)d_in[11];
    const float* kv_w  = (const float*)d_in[12];
    const float* ca_pw = (const float*)d_in[13];
    const float* ca_pb = (const float*)d_in[14];
    const float* ln3_g = (const float*)d_in[15];
    const float* ln3_b = (const float*)d_in[16];
    const float* fc1_w = (const float*)d_in[17];
    const float* fc1_b = (const float*)d_in[18];
    const float* fc2_w = (const float*)d_in[19];
    const float* fc2_b = (const float*)d_in[20];

    const int B = 4, N = 1024, C = 768, H = 12, Hd = 3072;
    const int M = B * N;

    char* ws = (char*)d_ws;
    size_t o = 0;
    auto alloc = [&](size_t bytes) -> void* {
        void* p = ws + o;
        o = (o + bytes + 255) & ~(size_t)255;
        return p;
    };
    float* x1    = (float*)alloc((size_t)M * C * 4);
    float* x2    = (float*)alloc((size_t)M * C * 4);
    u16*  h_bf   = (u16*)alloc((size_t)M * C * 2);
    u16*  qkv_bf = (u16*)alloc((size_t)M * 3 * C * 2);
    u16*  kv_bf  = (u16*)alloc((size_t)M * 2 * C * 2);
    u16*  q2_bf  = (u16*)alloc((size_t)M * C * 2);
    u16*  ao_bf  = (u16*)alloc((size_t)M * C * 2);
    u16*  enc_bf = (u16*)alloc((size_t)M * C * 2);
    u16*  fc1_bf = (u16*)alloc((size_t)M * Hd * 2);

    const dim3 gGemmC(C / GM_BN, M / GM_BM);
    const dim3 gGemm2C(2 * C / GM_BN, M / GM_BM);
    const dim3 gGemm3C(3 * C / GM_BN, M / GM_BM);
    const dim3 gGemmHd(Hd / GM_BN, M / GM_BM);
    const int  gAttn = B * H * (N / 64);

    // plain GEMM -> bf16 out           (qkv, q, kv)
    auto gemm_bf  = gemm_bf16_kernel<false, false, false, false, true>;
    // bias + residual -> fp32 out      (sa_proj, ca_proj, fc2)
    auto gemm_res = gemm_bf16_kernel<true, false, true, true, false>;
    // bias + GELU -> bf16 out          (fc1)
    auto gemm_gel = gemm_bf16_kernel<true, true, false, false, true>;

    // --- self attention ---
    ln_kernel<<<M, 256, 0, stream>>>(x, ln1_g, ln1_b, h_bf, C);
    gemm_bf<<<gGemm3C, 256, 0, stream>>>(h_bf, qkv_w, nullptr, nullptr,
                                         nullptr, qkv_bf, M, C, 3 * C);
    attn_kernel<<<gAttn, 128, 0, stream>>>(qkv_bf, 3 * C, qkv_bf, 3 * C, C, 2 * C,
                                           mask, ao_bf, C, H, N);
    gemm_res<<<gGemmC, 256, 0, stream>>>(ao_bf, sa_pw, sa_pb, x,
                                         x1, nullptr, M, C, C);
    // --- cross attention ---
    ln_kernel<<<M, 256, 0, stream>>>(x1, ln2_g, ln2_b, h_bf, C);
    gemm_bf<<<gGemmC, 256, 0, stream>>>(h_bf, q_w, nullptr, nullptr,
                                        nullptr, q2_bf, M, C, C);
    cvt_kernel<<<((long)M * C + 255) / 256, 256, 0, stream>>>(enc, enc_bf, (long)M * C);
    gemm_bf<<<gGemm2C, 256, 0, stream>>>(enc_bf, kv_w, nullptr, nullptr,
                                         nullptr, kv_bf, M, C, 2 * C);
    attn_kernel<<<gAttn, 128, 0, stream>>>(q2_bf, C, kv_bf, 2 * C, 0, C,
                                           emask, ao_bf, C, H, N);
    gemm_res<<<gGemmC, 256, 0, stream>>>(ao_bf, ca_pw, ca_pb, x1,
                                         x2, nullptr, M, C, C);
    // --- MLP ---
    ln_kernel<<<M, 256, 0, stream>>>(x2, ln3_g, ln3_b, h_bf, C);
    gemm_gel<<<gGemmHd, 256, 0, stream>>>(h_bf, fc1_w, fc1_b, nullptr,
                                          nullptr, fc1_bf, M, C, Hd);
    gemm_res<<<gGemmC, 256, 0, stream>>>(fc1_bf, fc2_w, fc2_b, x2,
                                         (float*)d_out, nullptr, M, Hd, C);
}
